// graph_encoder_pyg_1262720385763
// MI455X (gfx1250) — matrix operand order repaired and checked by reading, not compiled, not run on hardware
//
#include <hip/hip_runtime.h>
#include <hip/hip_bf16.h>
#include <math.h>

typedef __attribute__((ext_vector_type(16))) __bf16 v16bf;
typedef __attribute__((ext_vector_type(8)))  float  v8f;

#define F 128   // F_IN == F_HID == F_OUT == 128
// NOTE: M dims (100000 nodes, 256 graphs) are multiples of 16 -> no tile bounds checks.

// Branch-free tanh via hardware v_exp_f32 (libm tanhf is a multi-branch piecewise
// expansion that bloats the WMMA epilogue with s_and_saveexec blocks).
__device__ __forceinline__ float fast_tanh(float x) {
    x = fminf(fmaxf(x, -15.0f), 15.0f);     // keep exp in range; tanh saturated anyway
    float e = __expf(2.0f * x);
    return 1.0f - 2.0f / (e + 1.0f);
}

// ---------------- utility fills ----------------
__global__ void fill_kernel(float* p, float v, long long n) {
    long long i = (long long)blockIdx.x * blockDim.x + threadIdx.x;
    long long stride = (long long)gridDim.x * blockDim.x;
    for (; i < n; i += stride) p[i] = v;
}

// ---------------- degree / dinv ----------------
__global__ void degree_kernel(const int* col, float* deg, int E) {
    int e = blockIdx.x * blockDim.x + threadIdx.x;
    if (e < E) atomicAdd(&deg[col[e]], 1.0f);
}

__global__ void rsqrt_kernel(float* d, int n) {
    int i = blockIdx.x * blockDim.x + threadIdx.x;
    if (i < n) d[i] = rsqrtf(d[i]);   // deg >= 1 always (self-loop)
}

// ---------------- B-matrix pre-pack into WMMA lane layout (bf16) ----------------
// Bpack index = ((kt*8 + ntile)*32 + lane)*16 + i  where, per CDNA5 ISA 7.12.2
// (16-bit B 32x16): lane -> N = ntile*16 + (lane&15), K = kt*32 + (i>>3)*16 + (lane>>4)*8 + (i&7).
// One 32-byte vector load per fragment in the GEMM; whole pack is 32 KB (L0/L2 resident).
__global__ void pack_b_kernel(const float* __restrict__ W, __bf16* __restrict__ Bpack) {
    int idx  = blockIdx.x * blockDim.x + threadIdx.x;   // 0 .. 16383
    int i    = idx & 15;
    int lane = (idx >> 4) & 31;
    int nt   = (idx >> 9) & 7;
    int kt   = idx >> 12;
    int ncol = nt * 16 + (lane & 15);
    int k    = kt * 32 + (i >> 3) * 16 + (lane >> 4) * 8 + (i & 7);
    Bpack[idx] = (__bf16)W[(size_t)k * F + ncol];
}

// ---------------- WMMA GEMM: C[M x 128] = A[M x 128] @ B[128 x 128] ----------------
// Block = 256 threads = 8 waves; wave owns one 16-wide N tile, block owns a 16-row
// M tile. K=128 in 4 steps of v_wmma_f32_16x16x32_bf16 (f32 accumulate).
// A 16x32 bf16 fragment per lane = two contiguous 8-float runs:
//   a[0..7]  <- A[m0+(lane&15)][kbase + hi*8 .. +7]
//   a[8..15] <- A[m0+(lane&15)][kbase + 16 + hi*8 .. +7]      (hi = lane>>4)
// -> 4 aligned float4 loads / lane / K-step, casts fuse to v_cvt_pk_bf16_f32.
// C/D f32: vgpr r, lane -> M = r + 8*hi, N = lane&15.
__global__ void wmma_gemm_kernel(const float* __restrict__ A, const __bf16* __restrict__ Bpack,
                                 const float* __restrict__ bias, float* __restrict__ Cout,
                                 int mode) {
    const int wave = threadIdx.x >> 5;        // N tile 0..7
    const int lane = threadIdx.x & 31;
    const int m0   = blockIdx.x * 16;
    const int mr   = lane & 15;
    const int hi   = lane >> 4;
    const int ncol = wave * 16 + mr;

    const float* ap = A + (size_t)(m0 + mr) * F + hi * 8;
    const v16bf* bp = (const v16bf*)Bpack + wave * 32 + lane;

    v8f c = {};
    #pragma unroll
    for (int kt = 0; kt < 4; ++kt) {
        const float4 f0 = *(const float4*)(ap + kt * 32);
        const float4 f1 = *(const float4*)(ap + kt * 32 + 4);
        const float4 f2 = *(const float4*)(ap + kt * 32 + 16);
        const float4 f3 = *(const float4*)(ap + kt * 32 + 20);
        v16bf a;
        a[0]  = (__bf16)f0.x;  a[1]  = (__bf16)f0.y;
        a[2]  = (__bf16)f0.z;  a[3]  = (__bf16)f0.w;
        a[4]  = (__bf16)f1.x;  a[5]  = (__bf16)f1.y;
        a[6]  = (__bf16)f1.z;  a[7]  = (__bf16)f1.w;
        a[8]  = (__bf16)f2.x;  a[9]  = (__bf16)f2.y;
        a[10] = (__bf16)f2.z;  a[11] = (__bf16)f2.w;
        a[12] = (__bf16)f3.x;  a[13] = (__bf16)f3.y;
        a[14] = (__bf16)f3.z;  a[15] = (__bf16)f3.w;

        const v16bf b = bp[kt * 256];   // ((kt*8+wave)*32+lane) fragments, 32B vector load

        c = __builtin_amdgcn_wmma_f32_16x16x32_bf16(
                /*neg_a=*/false, a, /*neg_b=*/false, b,
                /*c_mod=*/(short)0, c, /*reuse_a=*/false, /*reuse_b=*/false);
    }

    #pragma unroll
    for (int r = 0; r < 8; ++r) {
        const int m = m0 + r + 8 * hi;
        float v = c[r];
        if (mode == 1) v = fast_tanh(v + bias[ncol]);
        Cout[(size_t)m * F + ncol] = v;
    }
}

// ---------------- edge gather + normalized scatter-add ----------------
// One wave per edge: 32 lanes x float4 = 128 features. h (51 MB) and agg (51 MB)
// both fit in MI455X's 192 MB L2, so gathers hit L2 and the fp32 atomics resolve
// in the L2 atomic units.
__global__ void edge_scatter_kernel(const int* __restrict__ row, const int* __restrict__ col,
                                    const float* __restrict__ dinv, const float* __restrict__ h,
                                    float* __restrict__ agg, int E) {
    const int e = blockIdx.x * (blockDim.x >> 5) + (threadIdx.x >> 5);
    if (e >= E) return;
    const int lane = threadIdx.x & 31;
    const int r  = row[e];
    const int cn = col[e];
    const float norm = dinv[r] * dinv[cn];
    const float4 v = ((const float4*)(h + (size_t)r * F))[lane];
    float* ap = agg + (size_t)cn * F + lane * 4;
    atomicAdd(ap + 0, v.x * norm);
    atomicAdd(ap + 1, v.y * norm);
    atomicAdd(ap + 2, v.z * norm);
    atomicAdd(ap + 3, v.w * norm);
}

// ---------------- bias + self-loop + ReLU + mean-pool accumulate ----------------
__global__ void relu_pool_kernel(const float* __restrict__ agg, const float* __restrict__ h,
                                 const float* __restrict__ dinv, const float* __restrict__ b1,
                                 const int* __restrict__ batch, float* __restrict__ pooled,
                                 float* __restrict__ cnt, int Nn) {
    const int node = blockIdx.x;
    if (node >= Nn) return;
    const int f = threadIdx.x;           // blockDim.x == 128
    const int g = batch[node];
    const float di = dinv[node];
    float v = agg[(size_t)node * F + f] + h[(size_t)node * F + f] * di * di + b1[f];
    v = fmaxf(v, 0.0f);
    atomicAdd(&pooled[(size_t)g * F + f], v);   // pooled: 128 KB, L2-resident
    if (f == 0) atomicAdd(&cnt[g], 1.0f);
}

__global__ void pool_norm_kernel(float* pooled, const float* cnt, int total) {
    int i = blockIdx.x * blockDim.x + threadIdx.x;
    if (i < total) pooled[i] /= fmaxf(cnt[i / F], 1.0f);
}

// ---------------- launch ----------------
extern "C" void kernel_launch(void* const* d_in, const int* in_sizes, int n_in,
                              void* d_out, int out_size, void* d_ws, size_t ws_size,
                              hipStream_t stream) {
    const float* x     = (const float*)d_in[0];
    const int*   ei    = (const int*)  d_in[1];   // [2, E] flat
    const int*   batch = (const int*)  d_in[2];
    const float* W1    = (const float*)d_in[3];
    const float* b1    = (const float*)d_in[4];
    const float* W2    = (const float*)d_in[5];
    const float* b2    = (const float*)d_in[6];
    float*       out   = (float*)d_out;

    const int Nn = in_sizes[2];          // 100000 nodes (multiple of 16)
    const int E  = in_sizes[1] / 2;      // 1600000 edges
    const int G  = out_size / F;         // 256 graphs (multiple of 16)

    // workspace carve-up (floats): dinv | h | agg | pooled | cnt | W1pack | W2pack
    float* ws     = (float*)d_ws;
    float* deg    = ws;                                   // Nn (becomes dinv)
    float* h      = ws + (size_t)((Nn + 255) & ~255);     // Nn*F
    float* agg    = h + (size_t)Nn * F;                   // Nn*F
    float* pooled = agg + (size_t)Nn * F;                 // G*F
    float* cnt    = pooled + (size_t)G * F;               // G
    __bf16* w1p   = (__bf16*)(cnt + ((G + 63) & ~63));    // 16384 bf16 (32-byte aligned)
    __bf16* w2p   = w1p + 16384;                          // 16384 bf16

    const int* row = ei;
    const int* col = ei + E;

    // init: deg = 1 (self-loop), agg = 0, pooled/cnt = 0
    fill_kernel<<<512, 256, 0, stream>>>(deg, 1.0f, (long long)Nn);
    fill_kernel<<<4096, 256, 0, stream>>>(agg, 0.0f, (long long)Nn * F);
    fill_kernel<<<64, 256, 0, stream>>>(pooled, 0.0f, (long long)G * F + G);

    // pack weights into WMMA B-fragment layout (bf16)
    pack_b_kernel<<<64, 256, 0, stream>>>(W1, w1p);
    pack_b_kernel<<<64, 256, 0, stream>>>(W2, w2p);

    // degree -> dinv
    degree_kernel<<<(E + 255) / 256, 256, 0, stream>>>(col, deg, E);
    rsqrt_kernel<<<(Nn + 255) / 256, 256, 0, stream>>>(deg, Nn);

    // h = x @ W1  (WMMA bf16, f32 accum)
    wmma_gemm_kernel<<<Nn / 16, 256, 0, stream>>>(x, w1p, nullptr, h, 0);

    // message passing: agg[col] += h[row] * norm
    edge_scatter_kernel<<<(E + 7) / 8, 256, 0, stream>>>(row, col, deg, h, agg, E);

    // bias + self-loop + relu + mean-pool accumulate
    relu_pool_kernel<<<Nn, F, 0, stream>>>(agg, h, deg, b1, batch, pooled, cnt, Nn);
    pool_norm_kernel<<<(G * F + 255) / 256, 256, 0, stream>>>(pooled, cnt, G * F);

    // out = tanh(pooled @ W2 + b2)  (WMMA, fused fast-tanh+bias epilogue)
    wmma_gemm_kernel<<<G / 16, 256, 0, stream>>>(pooled, w2p, b2, out, 1);
}
